// CoarsenRelu_28664611733896
// MI455X (gfx1250) — matrix operand order repaired and checked by reading, not compiled, not run on hardware
//
#include <hip/hip_runtime.h>

typedef __attribute__((ext_vector_type(16))) __bf16 v16bf;
typedef __attribute__((ext_vector_type(8)))  float  v8f;

#define N_COARSE 250000
#define KNB 9
#define C_IN 64
#define C_OUT 64
#define KDIM (KNB * C_IN)   /* 576 */
#define LDA  (KDIM + 8)     /* padded LDS row stride (bf16 elements) */

static __device__ __forceinline__ unsigned short f32_bf16(float f) {
  unsigned int u = __float_as_uint(f);
  u += 0x7fffu + ((u >> 16) & 1u);   // round-to-nearest-even
  return (unsigned short)(u >> 16);
}

// Prep: W fp32 [KDIM, C_OUT] row-major  ->  WT bf16 [C_OUT, KDIM] row-major
__global__ void wt_convert_kernel(const float* __restrict__ W,
                                  unsigned short* __restrict__ WT) {
  int t = blockIdx.x * blockDim.x + threadIdx.x;
  if (t >= KDIM * C_OUT) return;
  int k = t / C_OUT;
  int n = t - k * C_OUT;
  WT[n * KDIM + k] = f32_bf16(W[t]);
}

union ABits { uint4 u[2]; v16bf v; };

__global__ void __launch_bounds__(128)
coarsen_wmma_kernel(const float* __restrict__ lv,
                    const int*   __restrict__ nbr,
                    const unsigned short* __restrict__ WT,
                    const float* __restrict__ bias,
                    float* __restrict__ out) {
  __shared__ int shIdx[16 * KNB];
  __shared__ unsigned short shA[16 * LDA];

  const int tid = threadIdx.x;
  const int rowBase = blockIdx.x * 16;

  // Stage 0: neighbor indices for this 16-row tile into LDS
  if (tid < 16 * KNB) shIdx[tid] = nbr[rowBase * KNB + tid];
  __syncthreads();

  // Stage 1: gather fp32 segments (256B contiguous each), convert to bf16 in LDS.
  // 144 (row,nbr) segments x 16 float4 chunks = 2304 chunks / 128 threads.
  for (int c = tid; c < 16 * KNB * (C_IN / 4); c += 128) {
    int pair   = c >> 4;
    int within = (c & 15) * 4;
    int row = pair / KNB;
    int nb  = pair - row * KNB;
    const float4 f = *(const float4*)(lv + (long)shIdx[pair] * C_IN + within);
    unsigned int lo = (unsigned)f32_bf16(f.x) | ((unsigned)f32_bf16(f.y) << 16);
    unsigned int hi = (unsigned)f32_bf16(f.z) | ((unsigned)f32_bf16(f.w) << 16);
    uint2 packed; packed.x = lo; packed.y = hi;
    *(uint2*)&shA[row * LDA + nb * C_IN + within] = packed;
  }
  __syncthreads();

  // Stage 2: 4 waves, each computes a 16x16 N-tile over full K=576 (18 x K=32).
  const int wave  = tid >> 5;
  const int lane  = tid & 31;
  const int n     = lane & 15;
  const int kHalf = (lane < 16) ? 0 : 8;   // A layout: K 0-7/16-23 vs 8-15/24-31
  const int kOffB = (lane >> 4) * 8;       // B layout: K 0-7/16-23 vs 8-15/24-31

  const unsigned short* wrow = WT + (wave * 16 + n) * KDIM;
  const unsigned short* arow = &shA[(lane & 15) * LDA];

  v8f acc = {};
#pragma unroll
  for (int kk = 0; kk < KDIM / 32; ++kk) {
    ABits a, b;
    a.u[0] = *(const uint4*)(arow + kk * 32 + kHalf);        // ds_load_b128
    a.u[1] = *(const uint4*)(arow + kk * 32 + kHalf + 16);   // ds_load_b128
    b.u[0] = *(const uint4*)(wrow + kk * 32 + kOffB);        // global (L2-hit)
    b.u[1] = *(const uint4*)(wrow + kk * 32 + kOffB + 16);
    acc = __builtin_amdgcn_wmma_f32_16x16x32_bf16(false, a.v, false, b.v,
                                                  (short)0, acc, false, false);
  }

  // Epilogue: bias + ReLU + coalesced stores per C/D layout
  const float bn = bias[wave * 16 + n];
  const int mBase = (lane < 16) ? 0 : 8;
  float* op = out + (long)(rowBase + mBase) * C_OUT + wave * 16 + n;
#pragma unroll
  for (int r = 0; r < 8; ++r) {
    float v = acc[r] + bn;
    op[(long)r * C_OUT] = v > 0.f ? v : 0.f;
  }
}

extern "C" void kernel_launch(void* const* d_in, const int* in_sizes, int n_in,
                              void* d_out, int out_size, void* d_ws, size_t ws_size,
                              hipStream_t stream) {
  const float* lv  = (const float*)d_in[0];
  const int*   nbr = (const int*)d_in[1];
  const float* W   = (const float*)d_in[2];
  const float* b   = (const float*)d_in[3];
  float* out = (float*)d_out;
  unsigned short* WT = (unsigned short*)d_ws;   // 64*576*2 = 73,728 B scratch

  wt_convert_kernel<<<(KDIM * C_OUT + 255) / 256, 256, 0, stream>>>(W, WT);
  coarsen_wmma_kernel<<<N_COARSE / 16, 128, 0, stream>>>(lv, nbr, WT, b, out);
}
